// SelfAttentionBlock_37976100831505
// MI455X (gfx1250) — matrix operand order repaired and checked by reading, not compiled, not run on hardware
//
#include <hip/hip_runtime.h>
#include <hip/hip_bf16.h>
#include <math.h>

#define BB 4
#define SS 2048
#define DD 1024
#define NWAVES 8

typedef __attribute__((ext_vector_type(16))) __bf16 v16bf;
typedef __attribute__((ext_vector_type(8)))  __bf16 v8bf;
typedef __attribute__((ext_vector_type(8)))  float  v8f;
typedef __attribute__((ext_vector_type(4)))  float  v4f;

static __device__ __forceinline__ v16bf cat8(v8bf lo, v8bf hi) {
  return __builtin_shufflevector(lo, hi, 0,1,2,3,4,5,6,7,8,9,10,11,12,13,14,15);
}

// A-operand frag: 16x32 bf16 tile from row-major source (row = output row), ld elems.
// ISA layout: lanes 0-15 hold K = {0..7, 16..23}; lanes 16-31 hold K = {8..15, 24..31}.
// The two 16B halves are 32B apart -> two b128 loads + concat.
static __device__ __forceinline__ v16bf load_fragA(const __bf16* p, int ld, int lane) {
  int sub = lane & 15, hi = lane >> 4;
  const __bf16* rp = p + sub * ld + hi * 8;
  v8bf lo = *(const v8bf*)(rp);
  v8bf hh = *(const v8bf*)(rp + 16);
  return cat8(lo, hh);
}

// B-operand frag: 32x16 tile. Source row n holds B[:,n] contiguous over k (NT layout).
// Lane = column n; lanes 0-15 hold K = {0..7, 16..23}; lanes 16-31 hold K = {8..15, 24..31}
// (same order as the A frag) -> two b128 loads + concat.
static __device__ __forceinline__ v16bf load_fragB(const __bf16* p, int ld, int lane) {
  int sub = lane & 15, hi = lane >> 4;
  const __bf16* rp = p + sub * ld + hi * 8;
  v8bf lo = *(const v8bf*)(rp);
  v8bf hh = *(const v8bf*)(rp + 16);
  return cat8(lo, hh);
}

static __device__ __forceinline__ v8f wmma_bf16(v16bf a, v16bf b, v8f c) {
  return __builtin_amdgcn_wmma_f32_16x16x32_bf16(false, a, false, b, (short)0, c,
                                                 false, false);
}

// ---------------- Kernel 0a: convert x to bf16 (done once, reused 64x) -----------
__global__ void sa_pack_x(const float* __restrict__ x, __bf16* __restrict__ xb) {
  long i = ((long)blockIdx.x * blockDim.x + threadIdx.x) * 8;
  if (i >= (long)BB * SS * DD) return;
  v4f f0 = *(const v4f*)(x + i);
  v4f f1 = *(const v4f*)(x + i + 4);
  v8bf o;
#pragma unroll
  for (int j = 0; j < 4; ++j) {
    o[j] = (__bf16)f0[j];
    o[j + 4] = (__bf16)f1[j];
  }
  *(v8bf*)(xb + i) = o;
}

// ---------------- Kernel 0b: transpose + convert weights to bf16 -----------------
// Wt[w][n][k] = (bf16) W_w[k][n]
__global__ void sa_pack_weights(const float* __restrict__ Wq,
                                const float* __restrict__ Wk,
                                const float* __restrict__ Wv,
                                __bf16* __restrict__ Wt) {
  long idx = (long)blockIdx.x * blockDim.x + threadIdx.x;
  const long per = (long)DD * DD;
  if (idx >= 3 * per) return;
  int w = (int)(idx / per);
  long rem = idx - (long)w * per;
  int n = (int)(rem / DD);
  int k = (int)(rem % DD);
  const float* W = (w == 0) ? Wq : (w == 1) ? Wk : Wv;
  Wt[idx] = (__bf16)W[(long)k * DD + n];
}

// ---------------- Kernel 1: fused QKV projection GEMM (bf16 WMMA) ----------------
// One wave computes a 16x64 tile (4 accumulators). k-loop unrolled 2x with
// ping-pong fragment buffers: no inter-buffer copies, loads stay in flight.
__global__ __launch_bounds__(256) void sa_qkv_gemm(
    const __bf16* __restrict__ xb, const __bf16* __restrict__ Wt,
    const float* __restrict__ bq, const float* __restrict__ bk,
    const float* __restrict__ bv,
    __bf16* __restrict__ Qb, __bf16* __restrict__ Kb, __bf16* __restrict__ Vt) {
  const int lane = threadIdx.x & 31;
  const int wid = blockIdx.x * (blockDim.x >> 5) + (threadIdx.x >> 5);
  const int groups_per_mat = (BB * SS / 16) * (DD / 64);  // 8192
  int mat = wid / groups_per_mat;
  int t = wid - mat * groups_per_mat;
  int r0 = (t >> 4) << 4;  // row tile * 16 (rows flattened over B*S)
  int n0 = (t & 15) << 6;  // 64-wide output-column group
  const __bf16* Wm = Wt + (long)mat * DD * DD + (long)n0 * DD;
  const __bf16* arow = xb + (long)r0 * DD;
  const float* bias = (mat == 0) ? bq : (mat == 1) ? bk : bv;

  v8f acc[4] = {{}, {}, {}, {}};
  v16bf a0 = load_fragA(arow, DD, lane);
  v16bf b0[4];
#pragma unroll
  for (int j = 0; j < 4; ++j) b0[j] = load_fragB(Wm + (long)j * 16 * DD, DD, lane);

  for (int k = 0; k + 128 <= DD; k += 64) {
    v16bf a1 = load_fragA(arow + k + 32, DD, lane);
    v16bf b1[4];
#pragma unroll
    for (int j = 0; j < 4; ++j)
      b1[j] = load_fragB(Wm + (long)j * 16 * DD + k + 32, DD, lane);
#pragma unroll
    for (int j = 0; j < 4; ++j) acc[j] = wmma_bf16(a0, b0[j], acc[j]);
    a0 = load_fragA(arow + k + 64, DD, lane);
#pragma unroll
    for (int j = 0; j < 4; ++j)
      b0[j] = load_fragB(Wm + (long)j * 16 * DD + k + 64, DD, lane);
#pragma unroll
    for (int j = 0; j < 4; ++j) acc[j] = wmma_bf16(a1, b1[j], acc[j]);
  }
  {  // epilogue: chunks DD-64 (in a0/b0) and DD-32
    v16bf a1 = load_fragA(arow + DD - 32, DD, lane);
    v16bf b1[4];
#pragma unroll
    for (int j = 0; j < 4; ++j)
      b1[j] = load_fragB(Wm + (long)j * 16 * DD + DD - 32, DD, lane);
#pragma unroll
    for (int j = 0; j < 4; ++j) acc[j] = wmma_bf16(a0, b0[j], acc[j]);
#pragma unroll
    for (int j = 0; j < 4; ++j) acc[j] = wmma_bf16(a1, b1[j], acc[j]);
  }

  const int sub = lane & 15, hi = lane >> 4;
#pragma unroll
  for (int j = 0; j < 4; ++j) {
    int nc = n0 + j * 16;
    float bval = bias[nc + sub];
    if (mat < 2) {  // Q, K: row-major bf16 [B*S][D]
      __bf16* dst = (mat == 0) ? Qb : Kb;
#pragma unroll
      for (int r = 0; r < 8; ++r) {
        int m = hi * 8 + r;
        dst[(long)(r0 + m) * DD + nc + sub] = (__bf16)(acc[j][r] + bval);
      }
    } else {  // V: transposed [B][D][S]; 8 consecutive s per lane -> one 16B store
      int bI = r0 / SS;
      int s0 = r0 - bI * SS;
      v8bf pk;
#pragma unroll
      for (int r = 0; r < 8; ++r) pk[r] = (__bf16)(acc[j][r] + bval);
      *(v8bf*)(Vt + ((long)(bI * DD + nc + sub)) * SS + s0 + hi * 8) = pk;
    }
  }
}

// ---------------- Kernel 2: causal attention (scores + softmax + PV) ----------------
// One workgroup (8 waves) per 16-row query tile. Full score row kept in LDS as bf16.
__global__ __launch_bounds__(256) void sa_attn(const __bf16* __restrict__ Qb,
                                               const __bf16* __restrict__ Kb,
                                               const __bf16* __restrict__ Vt,
                                               float* __restrict__ out) {
  __shared__ __align__(32) __bf16 sQ[16 * DD];  // 32 KB, staged Q tile
  __shared__ __align__(32) __bf16 sP[16 * SS];  // 64 KB, scores then probs
  __shared__ float sRmax[NWAVES][16];
  __shared__ float sRsum[NWAVES][16];

  const int lane = threadIdx.x & 31;
  const int wave = threadIdx.x >> 5;
  const int sub = lane & 15, hi = lane >> 4;
  const int qt = blockIdx.x & (SS / 16 - 1);
  const int bI = blockIdx.x >> 7;  // / (SS/16)
  const int i0 = qt * 16;

  // stage the 16 x 1024 Q tile into LDS with CDNA5 async global->LDS copies
  {
    const char* src = (const char*)(Qb + ((long)bI * SS + i0) * DD);
    unsigned ldsbase = (unsigned)(unsigned long long)(const void*)sQ;  // addr[31:0] = LDS offset
    for (int i = threadIdx.x; i < 16 * DD * 2 / 16; i += blockDim.x) {
      unsigned lds = ldsbase + i * 16;
      unsigned long long ga = (unsigned long long)(src + i * 16);
      asm volatile("global_load_async_to_lds_b128 %0, %1, off"
                   :: "v"(lds), "v"(ga) : "memory");
    }
    asm volatile("s_wait_asynccnt 0" ::: "memory");
  }
  __syncthreads();

  const int ntiles = qt + 1;          // j-tiles up to and including the diagonal
  const int ntp = (ntiles + 1) & ~1;  // pad to even for K=32 PV chunks

  // ---- phase 1: scores S = (Q K^T)/sqrt(D), causal mask, track row max ----
  // Waves take PAIRS of j-tiles (shared Q A-frag); ping-pong pipelined; the
  // second tile is computed unconditionally (valid memory) and masked later.
  float vmax[8];
#pragma unroll
  for (int r = 0; r < 8; ++r) vmax[r] = -3.0e38f;

  for (int pt = wave; pt * 2 < ntp; pt += NWAVES) {
    int t0 = pt * 2, t1 = t0 + 1;
    v8f c0 = {}, c1 = {};
    const __bf16* kb0 = Kb + ((long)bI * SS + t0 * 16) * DD;
    const __bf16* kb1 = kb0 + 16 * DD;
    v16bf a0 = load_fragA(sQ, DD, lane);
    v16bf p0 = load_fragB(kb0, DD, lane);
    v16bf p1 = load_fragB(kb1, DD, lane);
    for (int k = 0; k + 128 <= DD; k += 64) {
      v16bf a1 = load_fragA(sQ + k + 32, DD, lane);
      v16bf q0 = load_fragB(kb0 + k + 32, DD, lane);
      v16bf q1 = load_fragB(kb1 + k + 32, DD, lane);
      c0 = wmma_bf16(a0, p0, c0);
      c1 = wmma_bf16(a0, p1, c1);
      a0 = load_fragA(sQ + k + 64, DD, lane);
      p0 = load_fragB(kb0 + k + 64, DD, lane);
      p1 = load_fragB(kb1 + k + 64, DD, lane);
      c0 = wmma_bf16(a1, q0, c0);
      c1 = wmma_bf16(a1, q1, c1);
    }
    {
      v16bf a1 = load_fragA(sQ + DD - 32, DD, lane);
      v16bf q0 = load_fragB(kb0 + DD - 32, DD, lane);
      v16bf q1 = load_fragB(kb1 + DD - 32, DD, lane);
      c0 = wmma_bf16(a0, p0, c0);
      c1 = wmma_bf16(a0, p1, c1);
      c0 = wmma_bf16(a1, q0, c0);
      c1 = wmma_bf16(a1, q1, c1);
    }

    bool v1 = (t1 < ntiles);
#pragma unroll
    for (int r = 0; r < 8; ++r) {
      int m = hi * 8 + r;
      float s0 = c0[r] * 0.03125f;  // 1/sqrt(1024)
      if (t0 == qt && sub > m) s0 = -1.0e30f;  // causal mask on diagonal tile
      vmax[r] = fmaxf(vmax[r], s0);
      sP[m * SS + t0 * 16 + sub] = (__bf16)s0;
      float s1 = -1.0e30f;
      if (v1) {
        s1 = c1[r] * 0.03125f;
        if (t1 == qt && sub > m) s1 = -1.0e30f;
        vmax[r] = fmaxf(vmax[r], s1);
      }
      sP[m * SS + t1 * 16 + sub] = (__bf16)s1;
    }
  }
  // per-row max within each 16-lane half (rows live per-VGPR per half)
#pragma unroll
  for (int r = 0; r < 8; ++r) {
    float v = vmax[r];
    v = fmaxf(v, __shfl_xor(v, 1, 32));
    v = fmaxf(v, __shfl_xor(v, 2, 32));
    v = fmaxf(v, __shfl_xor(v, 4, 32));
    v = fmaxf(v, __shfl_xor(v, 8, 32));
    vmax[r] = v;
  }
  if (sub == 0) {
#pragma unroll
    for (int r = 0; r < 8; ++r) sRmax[wave][hi * 8 + r] = vmax[r];
  }
  __syncthreads();

  float gmax[8];
#pragma unroll
  for (int r = 0; r < 8; ++r) {
    int m = hi * 8 + r;
    float v = sRmax[0][m];
#pragma unroll
    for (int w = 1; w < NWAVES; ++w) v = fmaxf(v, sRmax[w][m]);
    gmax[r] = v;
  }

  // ---- phase 2: P = exp(S - rowmax), accumulate row sums ----
  float vsum[8];
#pragma unroll
  for (int r = 0; r < 8; ++r) vsum[r] = 0.0f;
  for (int t = wave; t < ntp; t += NWAVES) {
#pragma unroll
    for (int r = 0; r < 8; ++r) {
      int m = hi * 8 + r;
      float s = (float)sP[m * SS + t * 16 + sub];
      float p = __expf(s - gmax[r]);
      vsum[r] += p;
      sP[m * SS + t * 16 + sub] = (__bf16)p;
    }
  }
#pragma unroll
  for (int r = 0; r < 8; ++r) {
    float v = vsum[r];
    v += __shfl_xor(v, 1, 32);
    v += __shfl_xor(v, 2, 32);
    v += __shfl_xor(v, 4, 32);
    v += __shfl_xor(v, 8, 32);
    vsum[r] = v;
  }
  if (sub == 0) {
#pragma unroll
    for (int r = 0; r < 8; ++r) sRsum[wave][hi * 8 + r] = vsum[r];
  }
  __syncthreads();  // also fences sP prob writes before PV reads

  float rinv[8];
#pragma unroll
  for (int r = 0; r < 8; ++r) {
    int m = hi * 8 + r;
    float v = sRsum[0][m];
#pragma unroll
    for (int w = 1; w < NWAVES; ++w) v += sRsum[w][m];
    rinv[r] = 1.0f / v;
  }

  // ---- phase 3: O = (P V) * rinv ; waves take groups of 4 d-tiles so each ----
  // ---- P A-frag from LDS feeds 4 WMMAs. Ping-pong pipelined Vt streams.   ----
  const int kkn = ntp >> 1;  // number of K=32 chunks over the j dimension
  for (int g = wave; g < DD / 64; g += NWAVES) {
    int d0 = g * 64;
    v8f acc[4] = {{}, {}, {}, {}};
    const __bf16* vb = Vt + ((long)bI * DD + d0) * SS;
    v16bf a0 = load_fragA(sP, SS, lane);
    v16bf b0[4];
#pragma unroll
    for (int j = 0; j < 4; ++j) b0[j] = load_fragB(vb + (long)j * 16 * SS, SS, lane);
    int kk = 1;
    for (; kk + 1 < kkn; kk += 2) {
      v16bf a1 = load_fragA(sP + kk * 32, SS, lane);
      v16bf b1[4];
#pragma unroll
      for (int j = 0; j < 4; ++j)
        b1[j] = load_fragB(vb + (long)j * 16 * SS + kk * 32, SS, lane);
#pragma unroll
      for (int j = 0; j < 4; ++j) acc[j] = wmma_bf16(a0, b0[j], acc[j]);
      a0 = load_fragA(sP + kk * 32 + 32, SS, lane);
#pragma unroll
      for (int j = 0; j < 4; ++j)
        b0[j] = load_fragB(vb + (long)j * 16 * SS + kk * 32 + 32, SS, lane);
#pragma unroll
      for (int j = 0; j < 4; ++j) acc[j] = wmma_bf16(a1, b1[j], acc[j]);
    }
    if (kk < kkn) {  // one more chunk to load, then drain both buffers
      v16bf a1 = load_fragA(sP + kk * 32, SS, lane);
      v16bf b1[4];
#pragma unroll
      for (int j = 0; j < 4; ++j)
        b1[j] = load_fragB(vb + (long)j * 16 * SS + kk * 32, SS, lane);
#pragma unroll
      for (int j = 0; j < 4; ++j) acc[j] = wmma_bf16(a0, b0[j], acc[j]);
#pragma unroll
      for (int j = 0; j < 4; ++j) acc[j] = wmma_bf16(a1, b1[j], acc[j]);
    } else {  // drain the last preloaded buffer
#pragma unroll
      for (int j = 0; j < 4; ++j) acc[j] = wmma_bf16(a0, b0[j], acc[j]);
    }

#pragma unroll
    for (int j = 0; j < 4; ++j) {
#pragma unroll
      for (int r = 0; r < 8; ++r) {
        int m = hi * 8 + r;
        out[((long)bI * SS + i0 + m) * DD + d0 + j * 16 + sub] = acc[j][r] * rinv[r];
      }
    }
  }
}

extern "C" void kernel_launch(void* const* d_in, const int* in_sizes, int n_in,
                              void* d_out, int out_size, void* d_ws, size_t ws_size,
                              hipStream_t stream) {
  const float* x  = (const float*)d_in[0];
  const float* Wq = (const float*)d_in[1];
  const float* bq = (const float*)d_in[2];
  const float* Wk = (const float*)d_in[3];
  const float* bk = (const float*)d_in[4];
  const float* Wv = (const float*)d_in[5];
  const float* bv = (const float*)d_in[6];
  float* out = (float*)d_out;

  // ws layout (bf16): Wt[3][D][D] | xb[B*S][D] | Qb[B*S][D] | Kb[B*S][D] | Vt[B][D][S]
  char* w = (char*)d_ws;
  const size_t WT = (size_t)3 * DD * DD * sizeof(__bf16);   // 6 MB
  const size_t MB = (size_t)BB * SS * DD * sizeof(__bf16);  // 16 MB each
  __bf16* Wt = (__bf16*)w;
  __bf16* xb = (__bf16*)(w + WT);
  __bf16* Qb = (__bf16*)(w + WT + MB);
  __bf16* Kb = (__bf16*)(w + WT + 2 * MB);
  __bf16* Vt = (__bf16*)(w + WT + 3 * MB);

  // kernel 0a: x -> bf16 (8 elems/thread)
  {
    long n = (long)BB * SS * DD / 8;
    sa_pack_x<<<(int)((n + 255) / 256), 256, 0, stream>>>(x, xb);
  }
  // kernel 0b: pack/transpose weights
  {
    long n = 3L * DD * DD;
    sa_pack_weights<<<(int)((n + 255) / 256), 256, 0, stream>>>(Wq, Wk, Wv, Wt);
  }
  // kernel 1: QKV projections (8 waves/block, 1 wave per 16x64 tile)
  {
    int waves = 3 * (BB * SS / 16) * (DD / 64);  // 24576
    sa_qkv_gemm<<<waves / 8, 256, 0, stream>>>(xb, Wt, bq, bk, bv, Qb, Kb, Vt);
  }
  // kernel 2: causal attention, one workgroup (8 waves) per 16-row query tile
  sa_attn<<<BB * (SS / 16), 256, 0, stream>>>(Qb, Kb, Vt, out);
}
